// Block_model_1700807049948
// MI455X (gfx1250) — matrix operand order repaired and checked by reading, not compiled, not run on hardware
//
#include <hip/hip_runtime.h>
#include <math.h>

// ---------------------------------------------------------------------------
// Problem constants
// ---------------------------------------------------------------------------
#define S_DIM 512
#define B_DIM 256
#define D_DIM 128
#define H_DIM 512

typedef __bf16 bf16;
typedef __attribute__((ext_vector_type(16))) bf16 v16bf;
typedef __attribute__((ext_vector_type(8)))  bf16 bf16x8;
typedef __attribute__((ext_vector_type(8)))  float v8f;

// ---------------------------------------------------------------------------
// WMMA helpers (wave32, 16x16x32 bf16 -> f32)
// A fragment: lane<16: row=lane,    k = k0+{0..7} and k0+{16..23}
//             lane>=16: row=lane-16, k = k0+{8..15} and k0+{24..31}
// B fragment: lane<16: col=lane,    k = k0+{0..7} and k0+{16..23}
//             lane>=16: col=lane-16, k = k0+{8..15} and k0+{24..31}
//   (B is loaded from a row-major [col][k] matrix, i.e. the transposed weight)
// C/D: lane holds col n=lane&15, rows m = (lane>=16?8:0)+r for vgpr r
// ---------------------------------------------------------------------------
static __device__ __forceinline__ v16bf frag_a(const bf16* base, int ld, int row0, int k0) {
  const int lane = threadIdx.x & 31;
  const bf16* p = base + (size_t)(row0 + (lane & 15)) * ld + (k0 + ((lane >> 4) & 1) * 8);
  bf16x8 lo = *(const bf16x8*)p;
  bf16x8 hi = *(const bf16x8*)(p + 16);
  v16bf r;
#pragma unroll
  for (int i = 0; i < 8; ++i) { r[i] = lo[i]; r[i + 8] = hi[i]; }
  return r;
}

static __device__ __forceinline__ v16bf frag_b(const bf16* base, int ld, int col0, int k0) {
  const int lane = threadIdx.x & 31;
  const bf16* p = base + (size_t)(col0 + (lane & 15)) * ld + (k0 + ((lane >> 4) & 1) * 8);
  bf16x8 lo = *(const bf16x8*)p;
  bf16x8 hi = *(const bf16x8*)(p + 16);
  v16bf r;
#pragma unroll
  for (int i = 0; i < 8; ++i) { r[i] = lo[i]; r[i + 8] = hi[i]; }
  return r;
}

static __device__ __forceinline__ v8f wmma_bf16(v16bf a, v16bf b, v8f c) {
  return __builtin_amdgcn_wmma_f32_16x16x32_bf16(false, a, false, b, (short)0, c, false, false);
}

static __device__ __forceinline__ void store_tile_bf16(bf16* base, int ld, int m0, int n0, v8f c) {
  const int lane = threadIdx.x & 31;
  const int n  = n0 + (lane & 15);
  const int mb = m0 + ((lane >> 4) & 1) * 8;
#pragma unroll
  for (int r = 0; r < 8; ++r) base[(size_t)(mb + r) * ld + n] = (bf16)c[r];
}

// ---------------------------------------------------------------------------
// Dynamic LDS layout (per attention block): 304128 bytes total (< 320KB WGP)
// ---------------------------------------------------------------------------
#define OFF_XS   0            // bf16 [512][128] : X, later overwritten by K
#define OFF_XT   131072       // bf16 [128][512] : X transposed (residual + P@X)
#define OFF_LNG  262144       // f32  [128]
#define OFF_LNB  262656       // f32  [128]
#define OFF_PB   263168       // bf16 per-wave [16][32]  exp(logits) staging
#define OFF_OB   271360       // bf16 per-wave [16][128] O / Q staging
#define SMEM_BYTES 304128

// ---------------------------------------------------------------------------
// Prep: bf16-convert (and transpose the [D,D] weights to [out][in]).
// ---------------------------------------------------------------------------
__global__ void __launch_bounds__(256)
prep_kernel(const float* __restrict__ Wq, const float* __restrict__ Wk, const float* __restrict__ Wv,
            const float* __restrict__ Wih, const float* __restrict__ Whh, const float* __restrict__ memf,
            bf16* __restrict__ wqT, bf16* __restrict__ wkT, bf16* __restrict__ wvT,
            bf16* __restrict__ wih, bf16* __restrict__ whh, bf16* __restrict__ memb)
{
  int i = blockIdx.x * blockDim.x + threadIdx.x;
  if (i < 32768) {                       // wqT[l][e][d] = Wq[l][d][e]
    int l = i >> 14, r = i & 16383, e = r >> 7, d = r & 127;
    wqT[i] = (bf16)Wq[(l << 14) + (d << 7) + e];
    return;
  }
  i -= 32768;
  if (i < 32768) {
    int l = i >> 14, r = i & 16383, e = r >> 7, d = r & 127;
    wkT[i] = (bf16)Wk[(l << 14) + (d << 7) + e];
    return;
  }
  i -= 32768;
  if (i < 32768) {
    int l = i >> 14, r = i & 16383, e = r >> 7, d = r & 127;
    wvT[i] = (bf16)Wv[(l << 14) + (d << 7) + e];
    return;
  }
  i -= 32768;
  if (i < 1536 * 2048) { wih[i] = (bf16)Wih[i]; return; }   // already [out][in]
  i -= 1536 * 2048;
  if (i < 1536 * 512)  { whh[i] = (bf16)Whh[i]; return; }   // already [out][in]
  i -= 1536 * 512;
  if (i < 256 * 512)   { memb[i] = (bf16)memf[i]; return; }
}

// ---------------------------------------------------------------------------
// Layer 1: full self-attention for one batch per workgroup (8 waves).
//   x1 = LN(x0 + softmax(QK^T/sqrt(D)) @ X0 @ Wv)
// ---------------------------------------------------------------------------
__global__ void __launch_bounds__(256)
attn_layer1(const float* __restrict__ x0,
            const bf16* __restrict__ wqT, const bf16* __restrict__ wkT, const bf16* __restrict__ wvT,
            const float* __restrict__ ln_g, const float* __restrict__ ln_b,
            bf16* __restrict__ wsQ, bf16* __restrict__ wsX1)
{
  extern __shared__ char smem[];
  bf16*  Xs  = (bf16*)(smem + OFF_XS);
  bf16*  XTs = (bf16*)(smem + OFF_XT);
  float* lng = (float*)(smem + OFF_LNG);
  float* lnb = (float*)(smem + OFF_LNB);
  const int b    = blockIdx.x;
  const int tid  = threadIdx.x;
  const int lane = tid & 31;
  const int wv   = tid >> 5;
  bf16* pb = (bf16*)(smem + OFF_PB) + wv * (16 * 32);
  bf16* ob = (bf16*)(smem + OFF_OB) + wv * (16 * 128);

  for (int i = tid; i < D_DIM; i += 256) { lng[i] = ln_g[i]; lnb[i] = ln_b[i]; }

  // stage x0 -> Xs (row major) and XTs (transposed), converting f32->bf16
  for (int idx = tid; idx < 512 * 32; idx += 256) {
    int s = idx >> 5, d4 = (idx & 31) << 2;
    float4 v = *(const float4*)(x0 + ((size_t)s * B_DIM + b) * D_DIM + d4);
    bf16 h0 = (bf16)v.x, h1 = (bf16)v.y, h2 = (bf16)v.z, h3 = (bf16)v.w;
    bf16* xr = Xs + (size_t)s * 128 + d4;
    xr[0] = h0; xr[1] = h1; xr[2] = h2; xr[3] = h3;
    XTs[(size_t)(d4 + 0) * 512 + s] = h0;
    XTs[(size_t)(d4 + 1) * 512 + s] = h1;
    XTs[(size_t)(d4 + 2) * 512 + s] = h2;
    XTs[(size_t)(d4 + 3) * 512 + s] = h3;
  }
  __syncthreads();

  const int row0w = wv * 64;
  bf16* Qb = wsQ + (size_t)b * (S_DIM * D_DIM);

  // Q = X @ Wq  -> global scratch (each wave owns 64 rows)
  for (int i = 0; i < 4; ++i) {
    int m0 = row0w + i * 16;
    v16bf a[4];
#pragma unroll
    for (int kk = 0; kk < 4; ++kk) a[kk] = frag_a(Xs, 128, m0, kk * 32);
#pragma unroll
    for (int j = 0; j < 8; ++j) {
      v8f c = {0, 0, 0, 0, 0, 0, 0, 0};
#pragma unroll
      for (int kk = 0; kk < 4; ++kk) c = wmma_bf16(a[kk], frag_b(wqT, 128, j * 16, kk * 32), c);
      store_tile_bf16(Qb, 128, m0, j * 16, c);
    }
  }
  // K = X @ Wk  -> in place over Xs (waves only touch their own rows)
  for (int i = 0; i < 4; ++i) {
    int m0 = row0w + i * 16;
    v16bf a[4];
#pragma unroll
    for (int kk = 0; kk < 4; ++kk) a[kk] = frag_a(Xs, 128, m0, kk * 32);
    v8f cc[8];
#pragma unroll
    for (int j = 0; j < 8; ++j) {
      v8f c = {0, 0, 0, 0, 0, 0, 0, 0};
#pragma unroll
      for (int kk = 0; kk < 4; ++kk) c = wmma_bf16(a[kk], frag_b(wkT, 128, j * 16, kk * 32), c);
      cc[j] = c;
    }
#pragma unroll
    for (int j = 0; j < 8; ++j) store_tile_bf16(Xs, 128, m0, j * 16, cc[j]);
  }
  __syncthreads();   // K fully resident in Xs

  const float scale = 0.08838834764831845f;   // 1/sqrt(128)
  for (int i = 0; i < 4; ++i) {
    int m0 = row0w + i * 16;
    v16bf qa[4];
#pragma unroll
    for (int kk = 0; kk < 4; ++kk) qa[kk] = frag_a(Qb, 128, m0, kk * 32);
    v8f ot[8];
#pragma unroll
    for (int j = 0; j < 8; ++j) ot[j] = (v8f){0, 0, 0, 0, 0, 0, 0, 0};
    float psum[8] = {0, 0, 0, 0, 0, 0, 0, 0};

    for (int t0 = 0; t0 < 512; t0 += 32) {
#pragma unroll
      for (int hlf = 0; hlf < 2; ++hlf) {
        int tt = t0 + hlf * 16;
        v8f c = {0, 0, 0, 0, 0, 0, 0, 0};
#pragma unroll
        for (int kk = 0; kk < 4; ++kk) c = wmma_bf16(qa[kk], frag_b(Xs, 128, tt, kk * 32), c);
        int mrow = ((lane >> 4) & 1) * 8;
        int tcol = hlf * 16 + (lane & 15);
#pragma unroll
        for (int r = 0; r < 8; ++r) {
          float e = __expf(c[r] * scale);      // no max-shift: logits are O(1)
          psum[r] += e;
          pb[(mrow + r) * 32 + tcol] = (bf16)e;
        }
      }
      // O^T += X^T[:, t0:t0+32] @ P^T   (A from XTs, B from row-major P)
#pragma unroll
      for (int j = 0; j < 8; ++j)
        ot[j] = wmma_bf16(frag_a(XTs, 512, j * 16, t0), frag_b(pb, 32, 0, 0), ot[j]);
    }

    // exp-rowsum: reduce across 16 lanes of each half, then broadcast per col m
#pragma unroll
    for (int r = 0; r < 8; ++r) {
      float s = psum[r];
      s += __shfl_xor(s, 1); s += __shfl_xor(s, 2); s += __shfl_xor(s, 4); s += __shfl_xor(s, 8);
      psum[r] = s;
    }
    float rs = 1.0f;
#pragma unroll
    for (int r = 0; r < 8; ++r) {
      float a0 = __shfl(psum[r], 0);
      float a1 = __shfl(psum[r], 16);
      if ((lane & 15) == r)     rs = a0;
      if ((lane & 15) == r + 8) rs = a1;
    }
    float inv = 1.0f / rs;

    // stage normalized O (=(A@X)) row-major for the @Wv GEMM
    {
      int m = lane & 15, bh = ((lane >> 4) & 1) * 8;
#pragma unroll
      for (int j = 0; j < 8; ++j)
#pragma unroll
        for (int r = 0; r < 8; ++r)
          ob[(size_t)m * 128 + j * 16 + bh + r] = (bf16)(ot[j][r] * inv);
    }
    v16bf oa[4];
#pragma unroll
    for (int kk = 0; kk < 4; ++kk) oa[kk] = frag_a(ob, 128, 0, kk * 32);
    v8f y[8];
#pragma unroll
    for (int j = 0; j < 8; ++j) {
      v8f c = {0, 0, 0, 0, 0, 0, 0, 0};
#pragma unroll
      for (int kk = 0; kk < 4; ++kk) c = wmma_bf16(oa[kk], frag_b(wvT, 128, j * 16, kk * 32), c);
      y[j] = c;
    }
    // residual + LayerNorm, write x1 (bf16, [B][S][D])
    const int n = lane & 15, bh = ((lane >> 4) & 1) * 8;
#pragma unroll
    for (int r = 0; r < 8; ++r) {
      int m = m0 + bh + r;
      float sum = 0.f, ssq = 0.f;
#pragma unroll
      for (int j = 0; j < 8; ++j) {
        int d = j * 16 + n;
        float v = y[j][r] + (float)XTs[(size_t)d * 512 + m];
        sum += v; ssq += v * v;
      }
      sum += __shfl_xor(sum, 1); sum += __shfl_xor(sum, 2); sum += __shfl_xor(sum, 4); sum += __shfl_xor(sum, 8);
      ssq += __shfl_xor(ssq, 1); ssq += __shfl_xor(ssq, 2); ssq += __shfl_xor(ssq, 4); ssq += __shfl_xor(ssq, 8);
      float mean = sum * (1.f / 128.f);
      float var  = ssq * (1.f / 128.f) - mean * mean;
      float rstd = rsqrtf(var + 1e-5f);
      bf16* dst = wsX1 + (size_t)b * (S_DIM * D_DIM) + (size_t)m * 128;
#pragma unroll
      for (int j = 0; j < 8; ++j) {
        int d = j * 16 + n;
        float v = y[j][r] + (float)XTs[(size_t)d * 512 + m];
        dst[d] = (bf16)((v - mean) * rstd * lng[d] + lnb[d]);
      }
    }
  }
}

// ---------------------------------------------------------------------------
// Layer 2: softmax rows sum to exactly 1 => top_k picks rows 0..15, so only
// 16 query rows are needed. K/V projections still span all 512 rows.
// Result rows go straight to the GRU "flat" input [B][16*128].
// ---------------------------------------------------------------------------
__global__ void __launch_bounds__(256)
attn_layer2(const bf16* __restrict__ x1,
            const bf16* __restrict__ wqT, const bf16* __restrict__ wkT, const bf16* __restrict__ wvT,
            const float* __restrict__ ln_g, const float* __restrict__ ln_b,
            bf16* __restrict__ wsFlat)
{
  extern __shared__ char smem[];
  bf16*  Xs  = (bf16*)(smem + OFF_XS);
  bf16*  XTs = (bf16*)(smem + OFF_XT);
  float* lng = (float*)(smem + OFF_LNG);
  float* lnb = (float*)(smem + OFF_LNB);
  const int b    = blockIdx.x;
  const int tid  = threadIdx.x;
  const int lane = tid & 31;
  const int wv   = tid >> 5;
  bf16* pb0 = (bf16*)(smem + OFF_PB);
  bf16* ob0 = (bf16*)(smem + OFF_OB);

  for (int i = tid; i < D_DIM; i += 256) { lng[i] = ln_g[i]; lnb[i] = ln_b[i]; }

  const bf16* src = x1 + (size_t)b * (S_DIM * D_DIM);
  for (int idx = tid; idx < 512 * 16; idx += 256) {
    int s = idx >> 4, d8 = (idx & 15) << 3;
    bf16x8 v = *(const bf16x8*)(src + (size_t)s * 128 + d8);
    *(bf16x8*)(Xs + (size_t)s * 128 + d8) = v;
#pragma unroll
    for (int e = 0; e < 8; ++e) XTs[(size_t)(d8 + e) * 512 + s] = v[e];
  }
  __syncthreads();

  // wave0: Q for rows 0..15 before its K strip overwrites them
  if (wv == 0) {
    v16bf a[4];
#pragma unroll
    for (int kk = 0; kk < 4; ++kk) a[kk] = frag_a(Xs, 128, 0, kk * 32);
#pragma unroll
    for (int j = 0; j < 8; ++j) {
      v8f c = {0, 0, 0, 0, 0, 0, 0, 0};
#pragma unroll
      for (int kk = 0; kk < 4; ++kk) c = wmma_bf16(a[kk], frag_b(wqT, 128, j * 16, kk * 32), c);
      store_tile_bf16(ob0, 128, 0, j * 16, c);
    }
  }
  // K = X @ Wk in place (all waves, own rows only)
  const int row0w = wv * 64;
  for (int i = 0; i < 4; ++i) {
    int m0 = row0w + i * 16;
    v16bf a[4];
#pragma unroll
    for (int kk = 0; kk < 4; ++kk) a[kk] = frag_a(Xs, 128, m0, kk * 32);
    v8f cc[8];
#pragma unroll
    for (int j = 0; j < 8; ++j) {
      v8f c = {0, 0, 0, 0, 0, 0, 0, 0};
#pragma unroll
      for (int kk = 0; kk < 4; ++kk) c = wmma_bf16(a[kk], frag_b(wkT, 128, j * 16, kk * 32), c);
      cc[j] = c;
    }
#pragma unroll
    for (int j = 0; j < 8; ++j) store_tile_bf16(Xs, 128, m0, j * 16, cc[j]);
  }
  __syncthreads();

  if (wv != 0) return;

  const float scale = 0.08838834764831845f;
  v16bf qa[4];
#pragma unroll
  for (int kk = 0; kk < 4; ++kk) qa[kk] = frag_a(ob0, 128, 0, kk * 32);
  v8f ot[8];
#pragma unroll
  for (int j = 0; j < 8; ++j) ot[j] = (v8f){0, 0, 0, 0, 0, 0, 0, 0};
  float psum[8] = {0, 0, 0, 0, 0, 0, 0, 0};

  for (int t0 = 0; t0 < 512; t0 += 32) {
#pragma unroll
    for (int hlf = 0; hlf < 2; ++hlf) {
      int tt = t0 + hlf * 16;
      v8f c = {0, 0, 0, 0, 0, 0, 0, 0};
#pragma unroll
      for (int kk = 0; kk < 4; ++kk) c = wmma_bf16(qa[kk], frag_b(Xs, 128, tt, kk * 32), c);
      int mrow = ((lane >> 4) & 1) * 8;
      int tcol = hlf * 16 + (lane & 15);
#pragma unroll
      for (int r = 0; r < 8; ++r) {
        float e = __expf(c[r] * scale);
        psum[r] += e;
        pb0[(mrow + r) * 32 + tcol] = (bf16)e;
      }
    }
#pragma unroll
    for (int j = 0; j < 8; ++j)
      ot[j] = wmma_bf16(frag_a(XTs, 512, j * 16, t0), frag_b(pb0, 32, 0, 0), ot[j]);
  }

#pragma unroll
  for (int r = 0; r < 8; ++r) {
    float s = psum[r];
    s += __shfl_xor(s, 1); s += __shfl_xor(s, 2); s += __shfl_xor(s, 4); s += __shfl_xor(s, 8);
    psum[r] = s;
  }
  float rs = 1.0f;
#pragma unroll
  for (int r = 0; r < 8; ++r) {
    float a0 = __shfl(psum[r], 0);
    float a1 = __shfl(psum[r], 16);
    if ((lane & 15) == r)     rs = a0;
    if ((lane & 15) == r + 8) rs = a1;
  }
  float inv = 1.0f / rs;

  {
    int m = lane & 15, bh = ((lane >> 4) & 1) * 8;
#pragma unroll
    for (int j = 0; j < 8; ++j)
#pragma unroll
      for (int r = 0; r < 8; ++r)
        ob0[(size_t)m * 128 + j * 16 + bh + r] = (bf16)(ot[j][r] * inv);
  }
  v16bf oa[4];
#pragma unroll
  for (int kk = 0; kk < 4; ++kk) oa[kk] = frag_a(ob0, 128, 0, kk * 32);
  v8f y[8];
#pragma unroll
  for (int j = 0; j < 8; ++j) {
    v8f c = {0, 0, 0, 0, 0, 0, 0, 0};
#pragma unroll
    for (int kk = 0; kk < 4; ++kk) c = wmma_bf16(oa[kk], frag_b(wvT, 128, j * 16, kk * 32), c);
    y[j] = c;
  }
  const int n = lane & 15, bh = ((lane >> 4) & 1) * 8;
  bf16* fb = wsFlat + (size_t)b * 2048;
#pragma unroll
  for (int r = 0; r < 8; ++r) {
    int m = bh + r;
    float sum = 0.f, ssq = 0.f;
#pragma unroll
    for (int j = 0; j < 8; ++j) {
      int d = j * 16 + n;
      float v = y[j][r] + (float)XTs[(size_t)d * 512 + m];
      sum += v; ssq += v * v;
    }
    sum += __shfl_xor(sum, 1); sum += __shfl_xor(sum, 2); sum += __shfl_xor(sum, 4); sum += __shfl_xor(sum, 8);
    ssq += __shfl_xor(ssq, 1); ssq += __shfl_xor(ssq, 2); ssq += __shfl_xor(ssq, 4); ssq += __shfl_xor(ssq, 8);
    float mean = sum * (1.f / 128.f);
    float var  = ssq * (1.f / 128.f) - mean * mean;
    float rstd = rsqrtf(var + 1e-5f);
#pragma unroll
    for (int j = 0; j < 8; ++j) {
      int d = j * 16 + n;
      float v = y[j][r] + (float)XTs[(size_t)d * 512 + m];
      fb[(size_t)m * 128 + d] = (bf16)((v - mean) * rstd * lng[d] + lnb[d]);
    }
  }
}

// ---------------------------------------------------------------------------
// GRU cell: gi = flat @ W_ih^T, gh = h @ W_hh^T, fused gates.
// One 16x16 output tile per wave; 128 blocks x 4 waves = 512 tiles.
// ---------------------------------------------------------------------------
__global__ void __launch_bounds__(128)
gru_kernel(const bf16* __restrict__ flat, const bf16* __restrict__ wih, const bf16* __restrict__ whh,
           const bf16* __restrict__ memb, const float* __restrict__ b_ih, const float* __restrict__ b_hh,
           const float* __restrict__ mem_f32, float* __restrict__ out)
{
  const int lane = threadIdx.x & 31;
  const int wv   = threadIdx.x >> 5;
  const int tile = blockIdx.x * 4 + wv;
  const int b0 = (tile >> 5) * 16;
  const int h0 = (tile & 31) * 16;

  v8f gr = {0,0,0,0,0,0,0,0}, gz = {0,0,0,0,0,0,0,0}, gn = {0,0,0,0,0,0,0,0};
  for (int kk = 0; kk < 64; ++kk) {
    v16bf a = frag_a(flat, 2048, b0, kk * 32);
    gr = wmma_bf16(a, frag_b(wih, 2048, h0,         kk * 32), gr);
    gz = wmma_bf16(a, frag_b(wih, 2048, 512 + h0,   kk * 32), gz);
    gn = wmma_bf16(a, frag_b(wih, 2048, 1024 + h0,  kk * 32), gn);
  }
  v8f hr = {0,0,0,0,0,0,0,0}, hz = {0,0,0,0,0,0,0,0}, hn = {0,0,0,0,0,0,0,0};
  for (int kk = 0; kk < 16; ++kk) {
    v16bf a = frag_a(memb, 512, b0, kk * 32);
    hr = wmma_bf16(a, frag_b(whh, 512, h0,        kk * 32), hr);
    hz = wmma_bf16(a, frag_b(whh, 512, 512 + h0,  kk * 32), hz);
    hn = wmma_bf16(a, frag_b(whh, 512, 1024 + h0, kk * 32), hn);
  }
  const int n = lane & 15, bh = ((lane >> 4) & 1) * 8;
  const int o = h0 + n;
  const float bir = b_ih[o], biz = b_ih[512 + o], bin = b_ih[1024 + o];
  const float bhr = b_hh[o], bhz = b_hh[512 + o], bhn = b_hh[1024 + o];
#pragma unroll
  for (int r = 0; r < 8; ++r) {
    int bi = b0 + bh + r;
    float rg = 1.f / (1.f + __expf(-(gr[r] + bir + hr[r] + bhr)));
    float zg = 1.f / (1.f + __expf(-(gz[r] + biz + hz[r] + bhz)));
    float ng = tanhf(gn[r] + bin + rg * (hn[r] + bhn));
    float hp = mem_f32[(size_t)bi * 512 + o];
    out[(size_t)bi * 512 + o] = (1.f - zg) * ng + zg * hp;
  }
}

// ---------------------------------------------------------------------------
// Host launcher. Workspace layout (bytes):
//   wsQ 0..32M | wsX1 32M..64M | wsFlat | wqT | wkT | wvT | wih | whh | memb
// Total ~73 MB.
// ---------------------------------------------------------------------------
extern "C" void kernel_launch(void* const* d_in, const int* in_sizes, int n_in,
                              void* d_out, int out_size, void* d_ws, size_t ws_size,
                              hipStream_t stream)
{
  (void)in_sizes; (void)n_in; (void)out_size; (void)ws_size;
  const float* x0   = (const float*)d_in[0];
  const float* mem  = (const float*)d_in[1];
  const float* Wq   = (const float*)d_in[2];
  const float* Wk   = (const float*)d_in[3];
  const float* Wv   = (const float*)d_in[4];
  const float* ln_g = (const float*)d_in[5];
  const float* ln_b = (const float*)d_in[6];
  const float* Wih  = (const float*)d_in[7];
  const float* Whh  = (const float*)d_in[8];
  const float* bih  = (const float*)d_in[9];
  const float* bhh  = (const float*)d_in[10];
  float* out = (float*)d_out;

  char* ws = (char*)d_ws;
  bf16* wsQ    = (bf16*)(ws + 0);
  bf16* wsX1   = (bf16*)(ws + 33554432);
  bf16* wsFlat = (bf16*)(ws + 67108864);
  bf16* wqT    = (bf16*)(ws + 68157440);
  bf16* wkT    = (bf16*)(ws + 68222976);
  bf16* wvT    = (bf16*)(ws + 68288512);
  bf16* wihb   = (bf16*)(ws + 68354048);
  bf16* whhb   = (bf16*)(ws + 74645504);
  bf16* memb   = (bf16*)(ws + 76218368);

  prep_kernel<<<16256, 256, 0, stream>>>(Wq, Wk, Wv, Wih, Whh, mem,
                                         wqT, wkT, wvT, wihb, whhb, memb);
  attn_layer1<<<B_DIM, 256, SMEM_BYTES, stream>>>(x0, wqT, wkT, wvT, ln_g, ln_b, wsQ, wsX1);
  attn_layer2<<<B_DIM, 256, SMEM_BYTES, stream>>>(wsX1, wqT + 16384, wkT + 16384, wvT + 16384,
                                                  ln_g + 128, ln_b + 128, wsFlat);
  gru_kernel<<<128, 128, 0, stream>>>(wsFlat, wihb, whhb, memb, bih, bhh, mem, out);
}
